// BinderDiffusionBlock_4063039062802
// MI455X (gfx1250) — matrix operand order repaired and checked by reading, not compiled, not run on hardware
//
#include <hip/hip_runtime.h>
#include <hip/hip_bf16.h>
#include <math.h>

// ---------------- problem constants ----------------
#define NN   2048
#define AA   5
#define DD   256
#define PP   128
#define HH   8
#define KT   32
#define KC   64
#define SIGMA_DATA 10.0f

// ---------------- input index map (setup_inputs dict order, leaves in insertion order)
enum {
  PF_T = 0, PF_C = 14,
  PF_W_REL=0, PF_W_TH=1, PF_W_DIST=2, PF_W_DIR=3, PF_W_ROT=4, PF_W_PVEC=5,
  PF_W_DIST_I=6, PF_W_ROT_I=7, PF_LN_S=8, PF_LN_B=9,
  PF_MLP_W1=10, PF_MLP_B1=11, PF_MLP_W2=12, PF_MLP_B2=13,
  AT_T = 28, AT_C = 34,
  AT_WQ=0, AT_WK=1, AT_WV=2, AT_WB=3, AT_WPV=4, AT_WO=5,
  UPD = 40,
  UP_W_COND=0, UP_MLP_W1=1, UP_MLP_B1=2, UP_MLP_W2=3, UP_MLP_B2=4,
  UP_W_UP=5, UP_W_GATE=6, UP_W_CGATE=7, UP_W_BGATE=8, UP_W_OUT=9, UP_B_OUT=10,
  LN1_S=51, LN1_B=52, LN2_S=53, LN2_B=54, LN3_S=55, LN3_B=56, LNF_S=57, LNF_B=58,
  W_POS=59, IN_LOCAL=60, IN_INCR=61, IN_POS=62, IN_PREV=63, IN_COND=64, IN_TIME=65,
  IN_ISTGT=66, IN_HOT=67, IN_RESI=68, IN_CHAIN=69, IN_BATCH=70, IN_MASK=71
};

// ---------------- bf16 / wmma plumbing ----------------
typedef __bf16 bf16_t;
typedef __attribute__((ext_vector_type(16))) __bf16 v16bf;
typedef __attribute__((ext_vector_type(8)))  float  v8f;

__device__ inline bf16_t f2bf(float x){ return (bf16_t)x; }
__device__ inline float  bf2f(bf16_t b){ return (float)b; }

// gelu(tanh approx) == x * sigmoid(1.5957691*(x+0.044715x^3)); branch-free via v_exp_f32
__device__ inline float gelu_f(float x){
  float z = 1.5957691216057308f * (x + 0.044715f * x * x * x);
  return x / (1.f + __expf(-z));
}
#define WMMA_BF16(a,b,c) __builtin_amdgcn_wmma_f32_16x16x32_bf16(false,(a),false,(b),(short)0,(c),false,false)

// A fragment (16x32 bf16): lane<16 -> row M=lane, K={0..7,16..23}; lane>=16 K shifted by 8.
// Direct-from-global: per lane two contiguous 8-element runs -> vectorizes, no LDS needed.
template<bool A_BF16>
__device__ inline v16bf load_A_global(const void* A, size_t row0, int ldk, int k0, int lane){
  int m = lane & 15, hi = lane >> 4;
  v16bf a;
  if (A_BF16){
    const bf16_t* Ar = (const bf16_t*)A + (row0 + m)*(size_t)ldk + k0 + hi*8;
#pragma unroll
    for (int j = 0; j < 8; ++j){ a[j] = Ar[j]; a[j+8] = Ar[16+j]; }
  } else {
    const float* Ar = (const float*)A + (row0 + m)*(size_t)ldk + k0 + hi*8;
#pragma unroll
    for (int j = 0; j < 8; ++j){ a[j] = f2bf(Ar[j]); a[j+8] = f2bf(Ar[16+j]); }
  }
  return a;
}
// A fragment from LDS bf16 tile (row-major, leading dim ld) -- two contiguous 8-elem runs
__device__ inline v16bf load_A_lds(const bf16_t* s, int ld, int k0, int lane){
  int m = lane & 15, hi = lane >> 4;
  const bf16_t* p = s + m*ld + k0 + hi*8;
  v16bf a;
#pragma unroll
  for (int j = 0; j < 8; ++j){ a[j] = p[j]; a[j+8] = p[16+j]; }
  return a;
}
// B fragment from PRE-PACKED weights: lane's 16 elements are contiguous (32B aligned)
// packed index = (((kc*CT + ct)*32 + lane)*16 + j), CT = Ncols/16
__device__ inline v16bf load_B_packed(const bf16_t* Bp, int CT, int kc, int ct, int lane){
  const bf16_t* p = Bp + (((size_t)(kc*CT + ct))*32 + lane)*16;
  v16bf b;
#pragma unroll
  for (int j = 0; j < 16; ++j) b[j] = p[j];
  return b;
}

// ---------------- fp32 weights -> bf16 WMMA-fragment-packed layout ----------------
// src: row-major [Kdim][Ncols] fp32. dst element i: j=i&15, lane=(i>>4)&31, tile=i>>9,
// kc=tile/CT, ct=tile%CT, k=kc*32 + (j&7) + 8*(lane>>4) + 16*(j>>3), c=ct*16 + (lane&15).
__global__ void pack_B_kernel(const float* __restrict__ src, bf16_t* __restrict__ dst,
                              int Kdim, int Ncols){
  int i = blockIdx.x*256 + threadIdx.x;
  if (i >= Kdim*Ncols) return;
  int CT = Ncols >> 4;
  int j = i & 15;
  int lane = (i >> 4) & 31;
  int tile = i >> 9;
  int kc = tile / CT, ct = tile - kc*CT;
  int k = kc*32 + (j & 7) + 8*(lane >> 4) + 16*(j >> 3);
  int c = ct*16 + (lane & 15);
  dst[i] = f2bf(src[(size_t)k*Ncols + c]);
}

// ---------------- geometry helpers ----------------
__device__ inline void make_frame(const float* pr, float E[3][3], float t[3]){
  float ca0=pr[3], ca1=pr[4], ca2=pr[5];
  float v0=pr[6]-ca0, v1=pr[7]-ca1, v2=pr[8]-ca2;
  float L = sqrtf(v0*v0+v1*v1+v2*v2) + 1e-8f;
  float e10=v0/L, e11=v1/L, e12=v2/L;
  float u0=pr[0]-ca0, u1=pr[1]-ca1, u2=pr[2]-ca2;
  float pj = u0*e10+u1*e11+u2*e12;
  float w0=u0-pj*e10, w1=u1-pj*e11, w2=u2-pj*e12;
  float L2 = sqrtf(w0*w0+w1*w1+w2*w2) + 1e-8f;
  float e20=w0/L2, e21=w1/L2, e22=w2/L2;
  E[0][0]=e10; E[0][1]=e11; E[0][2]=e12;
  E[1][0]=e20; E[1][1]=e21; E[1][2]=e22;
  E[2][0]=e11*e22-e12*e21; E[2][1]=e12*e20-e10*e22; E[2][2]=e10*e21-e11*e20;
  t[0]=ca0; t[1]=ca1; t[2]=ca2;
}

// ---------------- kNN (one workgroup per query row; stable argmin selection) ----------------
template<int K, bool TARGET>
__global__ void knn_kernel(const float* __restrict__ pos,
                           const int* __restrict__ resi, const int* __restrict__ chain,
                           const int* __restrict__ batch, const int* __restrict__ mask,
                           const int* __restrict__ is_target, const int* __restrict__ hotspots,
                           int* __restrict__ nbr)
{
  __shared__ float sd[NN];
  __shared__ float rmin[256];
  __shared__ int   ridx[256];
  const float FINF = 3.0e38f;
  int n = blockIdx.x, tid = threadIdx.x;
  int atom = TARGET ? 4 : 1;
  float qx = pos[n*15 + atom*3 + 0], qy = pos[n*15 + atom*3 + 1], qz = pos[n*15 + atom*3 + 2];
  int bn = batch[n]; int mn = mask[n];
  for (int j = tid; j < NN; j += 256){
    float dx = pos[j*15 + atom*3 + 0] - qx;
    float dy = pos[j*15 + atom*3 + 1] - qy;
    float dz = pos[j*15 + atom*3 + 2] - qz;
    float d = sqrtf(dx*dx + dy*dy + dz*dz);
    if (TARGET){
      if (!is_target[j]) d = FINF;
      if (hotspots[j])   d = 0.f;
    } else {
      int rd = resi[j] - resi[n]; rd = rd < 0 ? -rd : rd;
      if (chain[j] == chain[n] && rd <= 8) d = 0.f;
    }
    if (batch[j] != bn || !mask[j] || !mn) d = FINF;
    sd[j] = d;
  }
  __syncthreads();
  for (int s = 0; s < K; ++s){
    float best = FINF; int bidx = NN;
    for (int j = tid; j < NN; j += 256){
      float d = sd[j];
      if (d < best){ best = d; bidx = j; }
    }
    rmin[tid] = best; ridx[tid] = bidx;
    __syncthreads();
    for (int st = 128; st > 0; st >>= 1){
      if (tid < st){
        float a = rmin[tid], b = rmin[tid+st];
        if (b < a || (b == a && ridx[tid+st] < ridx[tid])){ rmin[tid]=b; ridx[tid]=ridx[tid+st]; }
      }
      __syncthreads();
    }
    if (tid == 0){
      int w = ridx[0];
      nbr[(size_t)n*K + s] = (rmin[0] >= FINF) ? -1 : w;
      if (w < NN) sd[w] = FINF;
    }
    __syncthreads();
  }
}

// ---------------- pair features + LayerNorm (one 128-thread block per (n,k)) ----------------
template<int K>
__global__ void pair_features_kernel(
    const float* __restrict__ pos, const float* __restrict__ prev, const int* __restrict__ nbr,
    const int* __restrict__ resi, const int* __restrict__ chain, const int* __restrict__ batch,
    const int* __restrict__ is_target, const int* __restrict__ hotspots, const int* __restrict__ mask,
    const float* __restrict__ w_rel, const float* __restrict__ w_th,
    const float* __restrict__ w_dist, const float* __restrict__ w_dir,
    const float* __restrict__ w_rot, const float* __restrict__ w_pvec,
    const float* __restrict__ w_dist_i, const float* __restrict__ w_rot_i,
    const float* __restrict__ ln_s, const float* __restrict__ ln_b,
    float* __restrict__ pair_out, float* __restrict__ pm_out)
{
  __shared__ float red[128];
  size_t nk = blockIdx.x;
  int n = (int)(nk / K);
  int p = threadIdx.x;
  int j = nbr[nk]; int jc = j < 0 ? 0 : j;

  int rd = resi[jc] - resi[n];
  rd = rd < -32 ? -32 : (rd > 32 ? 32 : rd); rd += 32;
  bool same = (chain[jc] == chain[n]) && (batch[jc] == batch[n]);
  int rel = same ? rd : 65;
  int an = is_target[n] ? 1 : 0, bj = is_target[jc] ? 1 : 0, hj = hotspots[jc] ? 1 : 0;
  float acc = w_rel[rel*PP + p] + w_th[(2*an + bj)*PP + p] + w_th[(4 + 2*an + hj)*PP + p];

  for (int s = 0; s < 2; ++s){
    const float* Ps = s ? prev : pos;
    const float* wd = s ? w_dist_i : w_dist;
    const float* wr = s ? w_rot_i : w_rot;
    float En[3][3], tn[3], Ej[3][3], tj[3];
    make_frame(Ps + (size_t)n*15, En, tn);
    make_frame(Ps + (size_t)jc*15, Ej, tj);
    float dx = Ps[jc*15+3]-Ps[n*15+3], dy = Ps[jc*15+4]-Ps[n*15+4], dz = Ps[jc*15+5]-Ps[n*15+5];
    float d = sqrtf(dx*dx + dy*dy + dz*dz);
#pragma unroll
    for (int i = 0; i < 16; ++i){
      float c = 22.f * (float)i / 15.f;
      float z = d - c;
      acc += __expf(-z*z * (1.f/3.78125f)) * wd[i*PP + p];
    }
    float f12[12];
#pragma unroll
    for (int a = 0; a < 3; ++a)
#pragma unroll
      for (int m = 0; m < 3; ++m)
        f12[a*3+m] = En[a][0]*Ej[m][0] + En[a][1]*Ej[m][1] + En[a][2]*Ej[m][2];
    float t0 = tj[0]-tn[0], t1 = tj[1]-tn[1], t2 = tj[2]-tn[2];
    float r0 = En[0][0]*t0+En[0][1]*t1+En[0][2]*t2;
    float r1 = En[1][0]*t0+En[1][1]*t1+En[1][2]*t2;
    float r2 = En[2][0]*t0+En[2][1]*t1+En[2][2]*t2;
    float Lr = sqrtf(r0*r0+r1*r1+r2*r2) + 1e-8f;
    f12[9]=r0/Lr; f12[10]=r1/Lr; f12[11]=r2/Lr;
#pragma unroll
    for (int i = 0; i < 12; ++i) acc += f12[i] * wr[i*PP + p];
  }

  {
    float En[3][3], tn[3];
    make_frame(pos + (size_t)n*15, En, tn);
    float dx = pos[jc*15+3]-pos[n*15+3], dy = pos[jc*15+4]-pos[n*15+4], dz = pos[jc*15+5]-pos[n*15+5];
    float L = sqrtf(dx*dx+dy*dy+dz*dz) + 1e-8f;
    float hx = dx/L, hy = dy/L, hz = dz/L;
#pragma unroll
    for (int a = 0; a < 3; ++a)
      acc += (En[a][0]*hx + En[a][1]*hy + En[a][2]*hz) * w_dir[a*PP + p];
#pragma unroll
    for (int at = 0; at < 5; ++at){
      float vx = pos[jc*15+at*3+0]-tn[0], vy = pos[jc*15+at*3+1]-tn[1], vz = pos[jc*15+at*3+2]-tn[2];
#pragma unroll
      for (int a = 0; a < 3; ++a)
        acc += (En[a][0]*vx + En[a][1]*vy + En[a][2]*vz) * w_pvec[(at*3+a)*PP + p];
    }
  }

  red[p] = acc; __syncthreads();
  for (int st = 64; st > 0; st >>= 1){ if (p < st) red[p] += red[p+st]; __syncthreads(); }
  float mu = red[0] / 128.f; __syncthreads();
  float c = acc - mu;
  red[p] = c*c; __syncthreads();
  for (int st = 64; st > 0; st >>= 1){ if (p < st) red[p] += red[p+st]; __syncthreads(); }
  float var = red[0] / 128.f;
  pair_out[nk*PP + p] = c * rsqrtf(var + 1e-5f) * ln_s[p] + ln_b[p];
  if (p == 0) pm_out[nk] = (mask[n] && mask[jc] && j != -1) ? 1.f : 0.f;
}

// ---------------- fused pair MLP: (16x128)@(128x256) gelu (16x256)@(256x128), WMMA bf16 ----------------
// w1 packed with Ncols=256 (CT=16), w2 packed with Ncols=128 (CT=8)
__global__ void pair_mlp_kernel(const float* __restrict__ pair_in,
                                const bf16_t* __restrict__ w1p, const float* __restrict__ b1,
                                const bf16_t* __restrict__ w2p, const float* __restrict__ b2,
                                bf16_t* __restrict__ pair_out)
{
  __shared__ bf16_t sH[16*256];
  size_t row0 = (size_t)blockIdx.x * 16;
  int wave = threadIdx.x >> 5, lane = threadIdx.x & 31;
  int nn = lane & 15, hi = lane >> 4;

  // GEMM1: each wave produces 32 hidden cols (two 16x16 tiles sharing each A fragment)
  {
    int c0 = wave*32, c1 = wave*32 + 16;
    int ct0 = wave*2, ct1 = wave*2 + 1;
    v8f acc0 = {}, acc1 = {};
#pragma unroll
    for (int kc = 0; kc < 4; ++kc){
      v16bf a = load_A_global<false>(pair_in, row0, PP, kc*32, lane);
      acc0 = WMMA_BF16(a, load_B_packed(w1p, 16, kc, ct0, lane), acc0);
      acc1 = WMMA_BF16(a, load_B_packed(w1p, 16, kc, ct1, lane), acc1);
    }
#pragma unroll
    for (int r = 0; r < 8; ++r){
      int m = r + 8*hi;
      sH[m*256 + c0 + nn] = f2bf(gelu_f(acc0[r] + b1[c0 + nn]));
      sH[m*256 + c1 + nn] = f2bf(gelu_f(acc1[r] + b1[c1 + nn]));
    }
  }
  __syncthreads();

  // GEMM2: each wave produces one 16x16 output tile from the LDS hidden tile
  {
    int col0 = wave*16;
    v8f acc = {};
#pragma unroll
    for (int kc = 0; kc < 8; ++kc){
      v16bf a = load_A_lds(sH, 256, kc*32, lane);
      acc = WMMA_BF16(a, load_B_packed(w2p, 8, kc, wave, lane), acc);
    }
#pragma unroll
    for (int r = 0; r < 8; ++r){
      int m = r + 8*hi;
      pair_out[(row0 + m)*PP + col0 + nn] = f2bf(acc[r] + b2[col0 + nn]);
    }
  }
}

// ---------------- generic WMMA GEMM (no LDS, no barriers, fragment-packed B) ----------------
// block = 256 threads = 8 waves; each wave computes a 16x32 output tile (2 WMMA tiles
// sharing one A fragment). grid = (M/16, Ncols/256). KDIM is compile-time.
template<int KDIM, bool A_BF16, bool OUT_BF16, bool GELU, bool ADDSRC, bool BIAS>
__global__ void gemm_kernel(const void* __restrict__ A, const bf16_t* __restrict__ Bp,
                            const float* __restrict__ bias, const float* __restrict__ addsrc,
                            void* __restrict__ Out, int Ncols)
{
  size_t row0 = (size_t)blockIdx.x * 16;
  int wave = threadIdx.x >> 5, lane = threadIdx.x & 31;
  int col0 = blockIdx.y*256 + wave*32;
  int CT = Ncols >> 4;
  int ct0 = col0 >> 4, ct1 = ct0 + 1;
  v8f acc0 = {}, acc1 = {};
#pragma unroll 4
  for (int k0 = 0; k0 < KDIM; k0 += 32){
    int kc = k0 >> 5;
    if (k0 + 32 < KDIM)
      __builtin_prefetch(&Bp[(((size_t)((kc+1)*CT + ct0))*32 + lane)*16], 0, 1);
    v16bf a = load_A_global<A_BF16>(A, row0, KDIM, k0, lane);
    acc0 = WMMA_BF16(a, load_B_packed(Bp, CT, kc, ct0, lane), acc0);
    acc1 = WMMA_BF16(a, load_B_packed(Bp, CT, kc, ct1, lane), acc1);
  }
  int nn = lane & 15, hi = lane >> 4;
#pragma unroll
  for (int t = 0; t < 2; ++t){
    v8f acc = t ? acc1 : acc0;
    int c0 = col0 + t*16;
#pragma unroll
    for (int r = 0; r < 8; ++r){
      int m = r + 8*hi;
      size_t o = (row0 + m)*(size_t)Ncols + c0 + nn;
      float v = acc[r];
      if (BIAS)   v += bias[c0 + nn];
      if (GELU)   v = gelu_f(v);
      if (ADDSRC) v += addsrc[o];
      if (OUT_BF16) ((bf16_t*)Out)[o] = f2bf(v);
      else          ((float*)Out)[o]  = v;
    }
  }
}

// ---------------- attention core (per-row: logits, softmax, weighted gather) ----------------
template<int K>
__global__ void attn_core_kernel(const float* __restrict__ q, const float* __restrict__ kbuf,
                                 const float* __restrict__ vbuf,
                                 const bf16_t* __restrict__ pair_bf, const bf16_t* __restrict__ pv_bf,
                                 const float* __restrict__ wb, const float* __restrict__ pm,
                                 const int* __restrict__ nbr, float* __restrict__ outbuf)
{
  __shared__ float sq[DD];
  __shared__ float sl[K*HH];
  __shared__ int   snbr[K];
  int n = blockIdx.x, tid = threadIdx.x;
  sq[tid] = q[(size_t)n*DD + tid];
  if (tid < K) snbr[tid] = nbr[(size_t)n*K + tid];
  __syncthreads();

  for (int t = tid; t < K*HH; t += 256){
    int kk = t >> 3, h = t & 7;
    int j = snbr[kk]; int jc = j < 0 ? 0 : j;
    float dot = 0.f;
#pragma unroll
    for (int i = 0; i < 32; ++i) dot += sq[h*32 + i] * kbuf[(size_t)jc*DD + h*32 + i];
    float pb = 0.f;
    const bf16_t* pr = pair_bf + ((size_t)n*K + kk)*PP;
#pragma unroll 16
    for (int p = 0; p < PP; ++p) pb += bf2f(pr[p]) * wb[p*HH + h];
    float l = dot * 0.17677669529663687f + pb;          // 1/sqrt(32)
    if (!(pm[(size_t)n*K + kk] > 0.f)) l = -1e9f;
    sl[t] = l;
  }
  __syncthreads();
  if (tid < HH){
    float mx = -1e30f;
    for (int kk = 0; kk < K; ++kk) mx = fmaxf(mx, sl[kk*HH + tid]);
    float s = 0.f;
    for (int kk = 0; kk < K; ++kk){ float e = __expf(sl[kk*HH + tid] - mx); sl[kk*HH + tid] = e; s += e; }
    float inv = 1.f / fmaxf(s, 1e-30f);
    for (int kk = 0; kk < K; ++kk) sl[kk*HH + tid] *= inv;
  }
  __syncthreads();
  {
    int h = tid >> 5;
    float acc = 0.f;
    for (int kk = 0; kk < K; ++kk){
      int j = snbr[kk]; int jc = j < 0 ? 0 : j;
      acc += sl[kk*HH + h] * (vbuf[(size_t)jc*DD + tid] + bf2f(pv_bf[((size_t)n*K + kk)*DD + tid]));
    }
    outbuf[(size_t)n*DD + tid] = acc;
  }
}

// ---------------- resi-dual: local = LN(local+upd), incr += upd ----------------
__global__ void resi_dual_kernel(float* __restrict__ local, float* __restrict__ incr,
                                 const float* __restrict__ upd,
                                 const float* __restrict__ s, const float* __restrict__ b)
{
  __shared__ float red[DD];
  int n = blockIdx.x, tid = threadIdx.x;
  size_t o = (size_t)n*DD + tid;
  float u = upd[o];
  float x = local[o] + u;
  incr[o] += u;
  red[tid] = x; __syncthreads();
  for (int st = 128; st > 0; st >>= 1){ if (tid < st) red[tid] += red[tid+st]; __syncthreads(); }
  float mu = red[0] / 256.f; __syncthreads();
  float c = x - mu;
  red[tid] = c*c; __syncthreads();
  for (int st = 128; st > 0; st >>= 1){ if (tid < st) red[tid] += red[tid+st]; __syncthreads(); }
  float var = red[0] / 256.f;
  local[o] = c * rsqrtf(var + 1e-5f) * s[tid] + b[tid];
}

// ---------------- update block helpers ----------------
__global__ void localpos_kernel(const float* __restrict__ pos, float* __restrict__ lp)
{
  int n = blockIdx.x*blockDim.x + threadIdx.x;
  if (n >= NN) return;
  float E[3][3], t[3];
  make_frame(pos + (size_t)n*15, E, t);
  for (int a = 0; a < AA; ++a){
    float vx = pos[n*15+a*3+0]-t[0], vy = pos[n*15+a*3+1]-t[1], vz = pos[n*15+a*3+2]-t[2];
    for (int j = 0; j < 3; ++j)
      lp[(size_t)n*16 + a*3 + j] = E[j][0]*vx + E[j][1]*vy + E[j][2]*vz;
  }
}
__global__ void mlp15_kernel(const float* __restrict__ lp, const float* __restrict__ w1,
                             const float* __restrict__ b1, float* __restrict__ h)
{
  size_t idx = (size_t)blockIdx.x*256 + threadIdx.x;    // < NN*512
  int n = (int)(idx >> 9), c = (int)(idx & 511);
  float s = b1[c];
#pragma unroll
  for (int a = 0; a < 15; ++a) s += lp[(size_t)n*16 + a] * w1[a*512 + c];
  h[idx] = gelu_f(s);
}
__global__ void segsum_kernel(const float* __restrict__ bg, const float* __restrict__ cg,
                              const float* __restrict__ lu,
                              const int* __restrict__ batch, const int* __restrict__ chain,
                              const int* __restrict__ mask,
                              float* __restrict__ sums, float* __restrict__ cnts)
{
  int seg = blockIdx.y;                                  // 0..7 batch, 8..23 chain
  int col = blockIdx.x*128 + threadIdx.x;
  bool isB = seg < 8;
  int segid = isB ? seg : seg - 8;
  const float* g = isB ? bg : cg;
  const int* idx = isB ? batch : chain;
  float acc = 0.f, cnt = 0.f;
  for (int n = 0; n < NN; ++n){
    if (idx[n] == segid){
      float mf = mask[n] ? 1.f : 0.f;
      acc += g[(size_t)n*512 + col] * lu[(size_t)n*512 + col] * mf;
      cnt += mf;
    }
  }
  sums[(size_t)seg*512 + col] = acc;
  if (blockIdx.x == 0 && threadIdx.x == 0) cnts[seg] = cnt;
}
__global__ void hidden_kernel(const float* __restrict__ lg, const float* __restrict__ lu,
                              const float* __restrict__ sums, const float* __restrict__ cnts,
                              const int* __restrict__ batch, const int* __restrict__ chain,
                              float* __restrict__ hidden)
{
  size_t idx = (size_t)blockIdx.x*256 + threadIdx.x;    // < NN*512
  int n = (int)(idx >> 9), c = (int)(idx & 511);
  int bs = batch[n], cs = 8 + chain[n];
  float h = lg[idx] * lu[idx];
  h += sums[(size_t)bs*512 + c] / fmaxf(cnts[bs], 1e-6f);
  h += sums[(size_t)cs*512 + c] / fmaxf(cnts[cs], 1e-6f);
  hidden[idx] = h;
}
__global__ void final_ln_kernel(const float* __restrict__ local, const float* __restrict__ incr,
                                const float* __restrict__ s, const float* __restrict__ b,
                                float* __restrict__ lnorm)
{
  __shared__ float red[DD];
  int n = blockIdx.x, tid = threadIdx.x;
  size_t o = (size_t)n*DD + tid;
  float x = incr[o];
  red[tid] = x; __syncthreads();
  for (int st = 128; st > 0; st >>= 1){ if (tid < st) red[tid] += red[tid+st]; __syncthreads(); }
  float mu = red[0] / 256.f; __syncthreads();
  float c = x - mu;
  red[tid] = c*c; __syncthreads();
  for (int st = 128; st > 0; st >>= 1){ if (tid < st) red[tid] += red[tid+st]; __syncthreads(); }
  float var = red[0] / 256.f;
  lnorm[o] = local[o] + c * rsqrtf(var + 1e-5f) * s[tid] + b[tid];
}
__global__ void posout_kernel(const float* __restrict__ pos, const float* __restrict__ lp,
                              const float* __restrict__ lnorm, const float* __restrict__ w_pos,
                              float* __restrict__ outpos)
{
  int n = blockIdx.x*blockDim.x + threadIdx.x;
  if (n >= NN) return;
  float u[15];
#pragma unroll
  for (int i = 0; i < 15; ++i){
    float s = 0.f;
    for (int d = 0; d < DD; ++d) s += lnorm[(size_t)n*DD + d] * w_pos[d*15 + i];
    u[i] = SIGMA_DATA * s;
  }
  float E[3][3], t[3];
  make_frame(pos + (size_t)n*15, E, t);
  for (int a = 0; a < AA; ++a){
    float y0 = lp[(size_t)n*16 + a*3 + 0] + u[a*3 + 0];
    float y1 = lp[(size_t)n*16 + a*3 + 1] + u[a*3 + 1];
    float y2 = lp[(size_t)n*16 + a*3 + 2] + u[a*3 + 2];
    for (int i = 0; i < 3; ++i)
      outpos[(size_t)n*15 + a*3 + i] = y0*E[0][i] + y1*E[1][i] + y2*E[2][i] + t[i];
  }
}

// ---------------- host-side pass driver ----------------
#define FIN(i) ((const float*)d_in[(i)])
#define IIN(i) ((const int*)d_in[(i)])

struct PassW {
  const bf16_t *w1, *w2;                 // pair MLP weights (packed bf16)
  const bf16_t *wq, *wk, *wv, *wpv, *wo; // attention weights (packed bf16)
};

template<int K>
static void run_pass(void* const* d_in, int pf, int at, const PassW& W, const int* nbr,
                     float* out_local, float* out_incr,
                     const float* lns, const float* lnb,
                     float* pair_f, bf16_t* pair_bf, bf16_t* pv_bf, float* pm,
                     float* qb, float* kb, float* vb, float* attnout, float* updbuf,
                     hipStream_t stream)
{
  const int NK = NN * K;
  pair_features_kernel<K><<<NK, 128, 0, stream>>>(
      FIN(IN_POS), FIN(IN_PREV), nbr,
      IIN(IN_RESI), IIN(IN_CHAIN), IIN(IN_BATCH), IIN(IN_ISTGT), IIN(IN_HOT), IIN(IN_MASK),
      FIN(pf+PF_W_REL), FIN(pf+PF_W_TH), FIN(pf+PF_W_DIST), FIN(pf+PF_W_DIR),
      FIN(pf+PF_W_ROT), FIN(pf+PF_W_PVEC), FIN(pf+PF_W_DIST_I), FIN(pf+PF_W_ROT_I),
      FIN(pf+PF_LN_S), FIN(pf+PF_LN_B), pair_f, pm);
  pair_mlp_kernel<<<NK/16, 256, 0, stream>>>(pair_f,
      W.w1, FIN(pf+PF_MLP_B1), W.w2, FIN(pf+PF_MLP_B2), pair_bf);
  gemm_kernel<DD,false,false,false,false,false><<<dim3(NN/16, 1), 256, 0, stream>>>(
      out_local, W.wq, nullptr, nullptr, qb, DD);
  gemm_kernel<DD,false,false,false,false,false><<<dim3(NN/16, 1), 256, 0, stream>>>(
      out_local, W.wk, nullptr, nullptr, kb, DD);
  gemm_kernel<DD,false,false,false,false,false><<<dim3(NN/16, 1), 256, 0, stream>>>(
      out_local, W.wv, nullptr, nullptr, vb, DD);
  gemm_kernel<PP,true,true,false,false,false><<<dim3(NK/16, 1), 256, 0, stream>>>(
      pair_bf, W.wpv, nullptr, nullptr, pv_bf, DD);
  attn_core_kernel<K><<<NN, 256, 0, stream>>>(qb, kb, vb, pair_bf, pv_bf,
      FIN(at+AT_WB), pm, nbr, attnout);
  gemm_kernel<DD,false,false,false,false,false><<<dim3(NN/16, 1), 256, 0, stream>>>(
      attnout, W.wo, nullptr, nullptr, updbuf, DD);
  resi_dual_kernel<<<NN, 256, 0, stream>>>(out_local, out_incr, updbuf, lns, lnb);
}

extern "C" void kernel_launch(void* const* d_in, const int* in_sizes, int n_in,
                              void* d_out, int out_size, void* d_ws, size_t ws_size,
                              hipStream_t stream)
{
  (void)in_sizes; (void)n_in; (void)out_size; (void)ws_size;
  char* base = (char*)d_ws;
  size_t off = 0;
  auto take = [&](size_t bytes) -> char* {
    char* p = base + off;
    off = (off + bytes + 255) & ~(size_t)255;
    return p;
  };

  int*    nbr_t   = (int*)    take((size_t)NN*KT*4);
  int*    nbr_c   = (int*)    take((size_t)NN*KC*4);
  float*  pair_f  = (float*)  take((size_t)NN*KC*PP*4);   // 64 MB, aliased by pv after MLP
  bf16_t* pv_bf   = (bf16_t*) pair_f;                     // NN*KC*DD*2 == same 64 MB
  float*  pm      = (float*)  take((size_t)NN*KC*4);
  bf16_t* pair_bf = (bf16_t*) take((size_t)NN*KC*PP*2);
  float*  qb      = (float*)  take((size_t)NN*DD*4);
  float*  kb      = (float*)  take((size_t)NN*DD*4);
  float*  vb      = (float*)  take((size_t)NN*DD*4);
  float*  attnout = (float*)  take((size_t)NN*DD*4);
  float*  updbuf  = (float*)  take((size_t)NN*DD*4);
  float*  h512    = (float*)  take((size_t)NN*512*4);
  float*  lu      = (float*)  take((size_t)NN*512*4);
  float*  lg      = (float*)  take((size_t)NN*512*4);
  float*  cg      = (float*)  take((size_t)NN*512*4);
  float*  bg      = (float*)  take((size_t)NN*512*4);
  float*  hidden  = (float*)  take((size_t)NN*512*4);
  float*  segsums = (float*)  take((size_t)24*512*4);
  float*  segcnt  = (float*)  take((size_t)24*4);
  float*  lp      = (float*)  take((size_t)NN*16*4);
  float*  lnorm   = (float*)  take((size_t)NN*DD*4);

  // weights pre-packed to bf16 WMMA-fragment layout (stay hot in L2)
  auto packW = [&](int idx, int Kdim, int Ncols) -> bf16_t* {
    int count = Kdim*Ncols;
    bf16_t* p = (bf16_t*)take((size_t)count*2);
    pack_B_kernel<<<(count+255)/256, 256, 0, stream>>>(FIN(idx), p, Kdim, Ncols);
    return p;
  };
  PassW Wt, Wc;
  Wt.w1  = packW(PF_T+PF_MLP_W1, PP, 2*PP);  Wt.w2 = packW(PF_T+PF_MLP_W2, 2*PP, PP);
  Wc.w1  = packW(PF_C+PF_MLP_W1, PP, 2*PP);  Wc.w2 = packW(PF_C+PF_MLP_W2, 2*PP, PP);
  Wt.wq  = packW(AT_T+AT_WQ, DD, DD);  Wt.wk  = packW(AT_T+AT_WK, DD, DD);
  Wt.wv  = packW(AT_T+AT_WV, DD, DD);  Wt.wpv = packW(AT_T+AT_WPV, PP, DD);
  Wt.wo  = packW(AT_T+AT_WO, DD, DD);
  Wc.wq  = packW(AT_C+AT_WQ, DD, DD);  Wc.wk  = packW(AT_C+AT_WK, DD, DD);
  Wc.wv  = packW(AT_C+AT_WV, DD, DD);  Wc.wpv = packW(AT_C+AT_WPV, PP, DD);
  Wc.wo  = packW(AT_C+AT_WO, DD, DD);
  bf16_t* cw_cond  = packW(UPD+UP_W_COND, DD, DD);
  bf16_t* cw_mlp2  = packW(UPD+UP_MLP_W2, 512, DD);
  bf16_t* cw_up    = packW(UPD+UP_W_UP,   DD, 512);
  bf16_t* cw_gate  = packW(UPD+UP_W_GATE, DD, 512);
  bf16_t* cw_cgate = packW(UPD+UP_W_CGATE,DD, 512);
  bf16_t* cw_bgate = packW(UPD+UP_W_BGATE,DD, 512);
  bf16_t* cw_out   = packW(UPD+UP_W_OUT,  512, DD);

  float* out_local = (float*)d_out;
  float* out_incr  = out_local + (size_t)NN*DD;
  float* out_pos   = out_incr  + (size_t)NN*DD;

  // local/incremental double as working buffers in d_out
  hipMemcpyAsync(out_local, d_in[IN_LOCAL], (size_t)NN*DD*4, hipMemcpyDeviceToDevice, stream);
  hipMemcpyAsync(out_incr,  d_in[IN_INCR],  (size_t)NN*DD*4, hipMemcpyDeviceToDevice, stream);

  // neighbourhoods
  knn_kernel<KT, true ><<<NN, 256, 0, stream>>>(FIN(IN_POS), IIN(IN_RESI), IIN(IN_CHAIN),
      IIN(IN_BATCH), IIN(IN_MASK), IIN(IN_ISTGT), IIN(IN_HOT), nbr_t);
  knn_kernel<KC, false><<<NN, 256, 0, stream>>>(FIN(IN_POS), IIN(IN_RESI), IIN(IN_CHAIN),
      IIN(IN_BATCH), IIN(IN_MASK), IIN(IN_ISTGT), IIN(IN_HOT), nbr_c);

  // pass 1: target neighbours (K=32), pass 2: current neighbours (K=64)
  run_pass<KT>(d_in, PF_T, AT_T, Wt, nbr_t, out_local, out_incr, FIN(LN1_S), FIN(LN1_B),
               pair_f, pair_bf, pv_bf, pm, qb, kb, vb, attnout, updbuf, stream);
  run_pass<KC>(d_in, PF_C, AT_C, Wc, nbr_c, out_local, out_incr, FIN(LN2_S), FIN(LN2_B),
               pair_f, pair_bf, pv_bf, pm, qb, kb, vb, attnout, updbuf, stream);

  // update block
  localpos_kernel<<<NN/256, 256, 0, stream>>>(FIN(IN_POS), lp);
  gemm_kernel<DD,false,false,false,true,false><<<dim3(NN/16, 1), 256, 0, stream>>>(
      FIN(IN_COND), cw_cond, nullptr, out_local, out_local, DD);
  mlp15_kernel<<<(NN*512)/256, 256, 0, stream>>>(lp, FIN(UPD+UP_MLP_W1), FIN(UPD+UP_MLP_B1), h512);
  gemm_kernel<512,false,false,false,true,true><<<dim3(NN/16, 1), 256, 0, stream>>>(
      h512, cw_mlp2, FIN(UPD+UP_MLP_B2), out_local, out_local, DD);
  gemm_kernel<DD,false,false,false,false,false><<<dim3(NN/16, 2), 256, 0, stream>>>(
      out_local, cw_up, nullptr, nullptr, lu, 512);
  gemm_kernel<DD,false,false,true,false,false><<<dim3(NN/16, 2), 256, 0, stream>>>(
      out_local, cw_gate, nullptr, nullptr, lg, 512);
  gemm_kernel<DD,false,false,true,false,false><<<dim3(NN/16, 2), 256, 0, stream>>>(
      out_local, cw_cgate, nullptr, nullptr, cg, 512);
  gemm_kernel<DD,false,false,true,false,false><<<dim3(NN/16, 2), 256, 0, stream>>>(
      out_local, cw_bgate, nullptr, nullptr, bg, 512);
  segsum_kernel<<<dim3(4, 24), 128, 0, stream>>>(bg, cg, lu,
      IIN(IN_BATCH), IIN(IN_CHAIN), IIN(IN_MASK), segsums, segcnt);
  hidden_kernel<<<(NN*512)/256, 256, 0, stream>>>(lg, lu, segsums, segcnt,
      IIN(IN_BATCH), IIN(IN_CHAIN), hidden);
  gemm_kernel<512,false,false,false,false,true><<<dim3(NN/16, 1), 256, 0, stream>>>(
      hidden, cw_out, FIN(UPD+UP_B_OUT), nullptr, updbuf, DD);
  resi_dual_kernel<<<NN, 256, 0, stream>>>(out_local, out_incr, updbuf, FIN(LN3_S), FIN(LN3_B));

  // final position update
  final_ln_kernel<<<NN, 256, 0, stream>>>(out_local, out_incr, FIN(LNF_S), FIN(LNF_B), lnorm);
  posout_kernel<<<NN/256, 256, 0, stream>>>(FIN(IN_POS), lp, lnorm, FIN(W_POS), out_pos);
}
